// SimplifiedSCNN_67233418051626
// MI455X (gfx1250) — matrix operand order repaired and checked by reading, not compiled, not run on hardware
//
#include <hip/hip_runtime.h>

// ---------------------------------------------------------------------------
// SimplifiedSCNN on MI455X (gfx1250, wave32, WMMA, async-to-LDS + TDM).
// zero -> prep weights (bf16 B-fragments) -> scatter -> conv1 (5x5, 1->32,
// K=32 pad, 2 WMMA / 16px) -> mask-down -> 3x stride-2 3x3 convs (32->32,
// 18 WMMA / 16px tile; inputs via global_load_async_to_lds_b128, weights via
// TENSOR_LOAD_TO_LDS) -> masked avg pool + MLP head.
// Activations bf16 (halves HBM traffic on dominant y1), f32 accumulation.
// ---------------------------------------------------------------------------

typedef __attribute__((ext_vector_type(8)))  float        v8f;
typedef __attribute__((ext_vector_type(16))) __bf16       v16bf;
typedef __attribute__((ext_vector_type(8)))  __bf16       v8bf;
typedef __attribute__((ext_vector_type(4)))  unsigned int v4u;
typedef __attribute__((ext_vector_type(4)))  unsigned int u32x4;
typedef __attribute__((ext_vector_type(8)))  int          i32x8;
typedef __attribute__((ext_vector_type(4)))  int          i32x4;

#define GB 8
#define GG 512
#define GC 32

__device__ __forceinline__ unsigned short f2bf_bits(float f) {
  unsigned u = __builtin_bit_cast(unsigned, f);
  u += 0x7FFFu + ((u >> 16) & 1u);       // round-to-nearest-even
  return (unsigned short)(u >> 16);
}
__device__ __forceinline__ float bf2f(unsigned short s) {
  return __builtin_bit_cast(float, ((unsigned)s) << 16);
}
__device__ __forceinline__ unsigned lds_off(const void* p) {
  return (unsigned)(unsigned long long)p;   // flat addr low 32b = LDS offset
}
__device__ __forceinline__ void async_b128(unsigned dst, unsigned long long src) {
  asm volatile("global_load_async_to_lds_b128 %0, %1, off"
               :: "v"(dst), "v"(src) : "memory");
}
__device__ __forceinline__ void async_b32(unsigned dst, unsigned long long src) {
  asm volatile("global_load_async_to_lds_b32 %0, %1, off"
               :: "v"(dst), "v"(src) : "memory");
}
__device__ __forceinline__ void wait_async0() {
  asm volatile("s_wait_asynccnt 0x0" ::: "memory");
}

// ---------------------------------------------------------------- zero ------
__global__ void k_zero(v4u* __restrict__ p, long n16) {
  long i  = (long)blockIdx.x * blockDim.x + threadIdx.x;
  long st = (long)gridDim.x * blockDim.x;
  v4u z = {};
  for (; i < n16; i += st) p[i] = z;
}

// ------------------------------------------------------------- scatter ------
__global__ void k_scatter(const int* __restrict__ coords,
                          const float* __restrict__ feats,
                          unsigned short* __restrict__ grid,
                          unsigned char* __restrict__ m1, int n) {
  int i = blockIdx.x * blockDim.x + threadIdx.x;
  if (i >= n) return;
  int b = coords[3 * i], y = coords[3 * i + 1], x = coords[3 * i + 2];
  size_t g = ((size_t)b * GG + y) * GG + x;
  grid[g] = f2bf_bits(feats[i]);
  m1[g]   = 1;
}

// -------------------------------------------- weight prep (bf16 B-frags) ---
// B-fragment (32x16 bf16, wave32): lane l holds column n=(l&15),
// elements j=0..15 -> K = (l<16 ? j : 16+j). Two halves for 32 out channels.
__global__ void k_prep(const float* __restrict__ K1, const float* __restrict__ K2,
                       const float* __restrict__ K3, const float* __restrict__ K4,
                       unsigned short* __restrict__ w1, unsigned short* __restrict__ w2,
                       unsigned short* __restrict__ w3, unsigned short* __restrict__ w4) {
  int i = blockIdx.x * blockDim.x + threadIdx.x;
  if (i < 1024) {                                      // K1: 25 taps -> K=32 pad
    int h = i >> 9, r = i & 511, l = r >> 4, j = r & 15;
    int n = (l & 15) + 16 * h;
    int k = (l < 16) ? j : 16 + j;
    float v = (k < 25) ? K1[k * 32 + n] : 0.f;
    w1[i] = f2bf_bits(v);
  }
  int i2 = i - 1024;
  if (i2 >= 0 && i2 < 3 * 9216) {                      // K2..K4: per-tap frags
    int which = i2 / 9216;
    int r0 = i2 - which * 9216;
    int tap = r0 >> 10;
    int r = r0 & 1023;
    int h = r >> 9; r &= 511;
    int l = r >> 4, j = r & 15;
    int n = (l & 15) + 16 * h;
    int k = (l < 16) ? j : 16 + j;
    const float* K = (which == 0) ? K2 : (which == 1 ? K3 : K4);
    unsigned short* W = (which == 0) ? w2 : (which == 1 ? w3 : w4);
    W[r0] = f2bf_bits(K[(tap * 32 + k) * 32 + n]);
  }
}

// -------------------------------------------------------- conv1 (WMMA) -----
// 5x5, 1->32ch, stride 1, pad 2, relu * m1. A = 16 px x 32 "taps" (25 real).
// Input tile (5 x 132 bf16) staged via async b32 (pixel pairs, 4B aligned).
__global__ void __launch_bounds__(256)
k_conv1(const unsigned short* __restrict__ grid,
        const unsigned char* __restrict__ m1,
        const unsigned short* __restrict__ w1,   // [2][32][16] lane-form
        unsigned short* __restrict__ y1) {
  __shared__ __align__(16) unsigned short tIn[5 * 132];
  const int tid  = threadIdx.x;
  const int wv   = tid >> 5;
  const int lane = tid & 31;
  const int x0   = blockIdx.x * 128;
  const int y    = blockIdx.y;
  const int b    = blockIdx.z;
  const unsigned short* gB = grid + (size_t)b * GG * GG;

  {
    const unsigned t0 = lds_off(&tIn[0]);
    for (int q = tid; q < 330; q += 256) {             // 5 rows x 66 px-pairs
      int r = q / 66, cc = q - r * 66;
      int iy = y + r - 2;
      int px = x0 - 2 + cc * 2;
      int o  = (r * 132 + cc * 2) * 2;
      if (iy >= 0 && iy < GG && px >= 0 && px < GG) {
        async_b32(t0 + (unsigned)o,
                  (unsigned long long)(const char*)gB +
                      (unsigned long long)((size_t)iy * GG + px) * 2ull);
      } else {
        *(unsigned*)((char*)&tIn[0] + o) = 0u;         // halo -> zero
      }
    }
  }
  v16bf b0 = *(const v16bf*)(w1 + (size_t)lane * 16);         // N=0..15
  v16bf b1 = *(const v16bf*)(w1 + (size_t)(32 + lane) * 16);  // N=16..31
  wait_async0();
  __syncthreads();

  const int m    = lane & 15;
  const int kAdd = (lane >> 4) << 4;   // A frag: lane>=16 holds K+16
  const int col0 = wv * 16 + m;

  v16bf a = {};
  #pragma unroll
  for (int j = 0; j < 16; ++j) {
    int k = j + kAdd;                  // K = {0..15} / {16..31}
    unsigned short raw = 0;
    if (k < 25) {
      int dy = k / 5, dx = k - dy * 5;
      raw = tIn[dy * 132 + col0 + dx];
    }
    a[j] = __builtin_bit_cast(__bf16, raw);
  }

  v8f acc0 = {}, acc1 = {};
  acc0 = __builtin_amdgcn_wmma_f32_16x16x32_bf16(false, a, false, b0, (short)0, acc0, false, false);
  acc1 = __builtin_amdgcn_wmma_f32_16x16x32_bf16(false, a, false, b1, (short)0, acc1, false, false);

  const int mrow = (lane >> 4) << 3;
  const int xw   = x0 + wv * 16;
  unsigned long long mv =
      *(const unsigned long long*)(m1 + ((size_t)b * GG + y) * GG + xw + mrow);
  unsigned short* oB = y1 + (((size_t)b * GG + y) * GG + xw) * GC;
  const int nlo = lane & 15;
  #pragma unroll
  for (int j = 0; j < 8; ++j) {
    const bool on = (mv >> (8 * j)) & 1ull;            // mask is 0/1 -> select
    float f0 = fmaxf(acc0[j], 0.f);
    float f1 = fmaxf(acc1[j], 0.f);
    f0 = on ? f0 : 0.f;
    f1 = on ? f1 : 0.f;
    size_t o = (size_t)(mrow + j) * GC + nlo;
    oB[o]      = f2bf_bits(f0);
    oB[o + 16] = f2bf_bits(f1);
  }
}

// ----------------------------------------------------- mask downsample -----
__global__ void k_down(const unsigned char* __restrict__ in,
                       unsigned char* __restrict__ out, int Ho, int Wo) {
  int i = blockIdx.x * blockDim.x + threadIdx.x;
  int n = GB * Ho * Wo;
  if (i >= n) return;
  int wo = i % Wo; int rest = i / Wo;
  int oy = rest % Ho; int b = rest / Ho;
  int Wi = Wo * 2;
  const unsigned char* p = in + (((size_t)b * Ho * 2) + 2 * oy) * Wi + 2 * wo;
  unsigned char v0 = p[0], v1 = p[1], v2 = p[Wi], v3 = p[Wi + 1];
  unsigned char v = v0 > v1 ? v0 : v1;
  v = v > v2 ? v : v2; v = v > v3 ? v : v3;
  out[i] = v;
}

// ------------------------------------------- stride-2 3x3 conv (WMMA) ------
// 32->32ch, pad 1, relu * mask. Block = 64 output px of one row (4 waves x
// 16 px). Weights (18KB, contiguous) staged via one TENSOR_LOAD_TO_LDS (TDM);
// input rows (3 x 129 px x 32ch bf16) via global_load_async_to_lds_b128.
// Inner loop is pure ds_load_b128 + v_wmma.
__global__ void __launch_bounds__(128)
k_conv_s2(const unsigned short* __restrict__ in,
          unsigned short* __restrict__ out,
          const unsigned char* __restrict__ mOut,
          const unsigned short* __restrict__ w,   // [9][2][32][16] lane-form
          int Hin, int Win, int Hout, int Wout) {
  __shared__ __align__(16) unsigned short sIn[3 * 130 * 32];  // 24960 B
  __shared__ __align__(32) unsigned short wLds[9 * 1024];     // 18432 B
  const int tid = threadIdx.x;
  const int x0  = blockIdx.x * 64;
  const int oy  = blockIdx.y;
  const int b   = blockIdx.z;
  const unsigned short* inB = in + (size_t)b * Hin * Win * GC;
  const int ixBase = 2 * x0 - 1;

  if (tid < 32) {                                      // wave 0: TDM weight DMA
    const unsigned long long ga = (unsigned long long)(const void*)w;
    const unsigned ldsA = lds_off(&wLds[0]);
    // D# group0: count=1 | lds_addr | global_addr[56:0] | type=2
    u32x4 g0 = { 1u, ldsA, (unsigned)ga,
                 (unsigned)((ga >> 32) & 0x01FFFFFFull) | 0x80000000u };
    // D# group1: data_size=2B; tensor_dim0=tile_dim0=9216; tensor_dim1=1;
    // dim0_stride=9216; no multicast/pad/iterate/barrier.
    i32x8 g1 = { (int)(1u << 16),
                 (int)((9216u & 0xFFFFu) << 16),
                 (int)((9216u >> 16) | (1u << 16)),
                 (int)(9216u << 16),
                 0,
                 (int)9216,
                 0, 0 };
    i32x4 gz4 = { 0, 0, 0, 0 };
    i32x8 gz8 = { 0, 0, 0, 0, 0, 0, 0, 0 };
    __builtin_amdgcn_tensor_load_to_lds(g0, g1, gz4, gz4, gz8, 0);
    __builtin_amdgcn_s_wait_tensorcnt(0);
  }
  {                                                    // input rows -> LDS
    const unsigned s0 = lds_off(&sIn[0]);
    const unsigned long long gi = (unsigned long long)(const char*)inB;
    for (int q = tid; q < 1548; q += 128) {            // 3 * 129 * 4 chunks
      int r = q / 516, rem = q - r * 516;
      int p = rem >> 2, c = rem & 3;
      int iy = 2 * oy - 1 + r;
      int ix = ixBase + p;
      int o  = (r * 130 + p) * 64 + c * 16;
      if (iy >= 0 && iy < Hin && ix >= 0 && ix < Win) {
        async_b128(s0 + (unsigned)o,
                   gi + (unsigned long long)((size_t)iy * Win + ix) * 64ull +
                       (unsigned long long)(c * 16));
      } else {
        v4u z = {};
        *(v4u*)((char*)&sIn[0] + o) = z;               // halo -> zero
      }
    }
  }
  wait_async0();
  __syncthreads();

  const int wv   = tid >> 5, lane = tid & 31;
  const int m    = lane & 15;
  const int cLo  = (lane >> 4) << 4;                   // A frag channel base
  const int pCol = 2 * (wv * 16 + m);                  // local input px base

  v8f acc0 = {}, acc1 = {};
  #pragma unroll
  for (int ky = 0; ky < 3; ++ky) {
    #pragma unroll
    for (int kx = 0; kx < 3; ++kx) {
      const unsigned short* ap = sIn + ((size_t)ky * 130 + pCol + kx) * 32 + cLo;
      v8bf lo = *(const v8bf*)(ap);                    // ds_load_b128
      v8bf hi = *(const v8bf*)(ap + 8);                // ds_load_b128
      v16bf a = __builtin_shufflevector(lo, hi, 0, 1, 2, 3, 4, 5, 6, 7,
                                        8, 9, 10, 11, 12, 13, 14, 15);
      const int tap = ky * 3 + kx;
      const v16bf b0 = *(const v16bf*)(wLds + (tap * 2 + 0) * 512 + lane * 16);
      const v16bf b1 = *(const v16bf*)(wLds + (tap * 2 + 1) * 512 + lane * 16);
      acc0 = __builtin_amdgcn_wmma_f32_16x16x32_bf16(false, a, false, b0, (short)0, acc0, false, false);
      acc1 = __builtin_amdgcn_wmma_f32_16x16x32_bf16(false, a, false, b1, (short)0, acc1, false, false);
    }
  }

  const int mrow = (lane >> 4) << 3;
  const int oxw  = x0 + wv * 16;
  unsigned long long mv =
      *(const unsigned long long*)(mOut + ((size_t)b * Hout + oy) * Wout + oxw + mrow);
  unsigned short* oB = out + (((size_t)b * Hout + oy) * Wout + oxw) * GC;
  const int nlo = lane & 15;
  #pragma unroll
  for (int j = 0; j < 8; ++j) {
    const bool on = (mv >> (8 * j)) & 1ull;
    float f0 = fmaxf(acc0[j], 0.f);
    float f1 = fmaxf(acc1[j], 0.f);
    f0 = on ? f0 : 0.f;
    f1 = on ? f1 : 0.f;
    size_t o = (size_t)(mrow + j) * GC + nlo;
    oB[o]      = f2bf_bits(f0);
    oB[o + 16] = f2bf_bits(f1);
  }
}

// ------------------------------------------------- pool + MLP head ---------
__global__ void __launch_bounds__(256)
k_head(const unsigned short* __restrict__ y4, const unsigned char* __restrict__ m4,
       const float* __restrict__ x2,
       const float* __restrict__ W1, const float* __restrict__ b1,
       const float* __restrict__ W2, const float* __restrict__ b2,
       const float* __restrict__ W3, const float* __restrict__ b3,
       float* __restrict__ out) {
  const int b = blockIdx.x, tid = threadIdx.x;
  __shared__ float red[256];
  __shared__ float z[64];
  __shared__ float h1[64];
  __shared__ float cntS;

  float c = 0.f;
  for (int p = tid; p < 4096; p += 256) c += (float)m4[b * 4096 + p];
  red[tid] = c; __syncthreads();
  for (int s = 128; s > 0; s >>= 1) { if (tid < s) red[tid] += red[tid + s]; __syncthreads(); }
  if (tid == 0) cntS = red[0] > 1.f ? red[0] : 1.f;
  __syncthreads();

  const int ch = tid & 31, grp = tid >> 5;
  float s = 0.f;
  const unsigned short* yB = y4 + (size_t)b * 4096 * 32;
  for (int p = grp; p < 4096; p += 8) s += bf2f(yB[(size_t)p * 32 + ch]);
  red[tid] = s; __syncthreads();
  if (tid < 32) {
    float tot = 0.f;
    for (int g = 0; g < 8; ++g) tot += red[g * 32 + tid];
    z[tid] = tot / cntS;                       // pooled -> z[0..31]
  }
  if (tid < 64) {
    float acc = b1[tid];
    for (int k = 0; k < 3; ++k) acc += x2[b * 3 + k] * W1[k * 64 + tid];
    h1[tid] = acc > 0.f ? acc : 0.f;
  }
  __syncthreads();
  if (tid < 32) {
    float acc = b2[tid];
    for (int k = 0; k < 64; ++k) acc += h1[k] * W2[k * 32 + tid];
    z[32 + tid] = acc;                          // h -> z[32..63]
  }
  __syncthreads();
  if (tid < 128) {
    float acc = b3[tid];
    for (int k = 0; k < 64; ++k) acc += z[k] * W3[k * 128 + tid];
    out[b * 128 + tid] = acc > 0.f ? acc : 0.f;
  }
}

// ---------------------------------------------------------------------------
extern "C" void kernel_launch(void* const* d_in, const int* in_sizes, int n_in,
                              void* d_out, int out_size, void* d_ws, size_t ws_size,
                              hipStream_t stream) {
  (void)in_sizes; (void)n_in; (void)out_size; (void)ws_size;
  const int*   coords = (const int*)  d_in[0];
  const float* feats  = (const float*)d_in[1];
  const float* x2     = (const float*)d_in[2];
  const float* K1     = (const float*)d_in[3];
  const float* K2     = (const float*)d_in[4];
  const float* K3     = (const float*)d_in[5];
  const float* K4     = (const float*)d_in[6];
  const float* W1     = (const float*)d_in[7];
  const float* b1     = (const float*)d_in[8];
  const float* W2     = (const float*)d_in[9];
  const float* b2     = (const float*)d_in[10];
  const float* W3     = (const float*)d_in[11];
  const float* b3     = (const float*)d_in[12];
  float* out = (float*)d_out;

  char* ws = (char*)d_ws;
  size_t off = 0;
  auto take = [&](size_t bytes) {
    size_t o = off; off += (bytes + 255) & ~(size_t)255; return o;
  };
  unsigned short* grid = (unsigned short*)(ws + take((size_t)GB * GG * GG * 2));
  unsigned char*  m1   = (unsigned char*) (ws + take((size_t)GB * GG * GG));
  unsigned short* y1   = (unsigned short*)(ws + take((size_t)GB * GG * GG * GC * 2));
  unsigned char*  m2   = (unsigned char*) (ws + take((size_t)GB * 256 * 256));
  unsigned short* y2   = (unsigned short*)(ws + take((size_t)GB * 256 * 256 * GC * 2));
  unsigned char*  m3   = (unsigned char*) (ws + take((size_t)GB * 128 * 128));
  unsigned short* y3   = (unsigned short*)(ws + take((size_t)GB * 128 * 128 * GC * 2));
  unsigned char*  m4   = (unsigned char*) (ws + take((size_t)GB * 64 * 64));
  unsigned short* y4   = (unsigned short*)(ws + take((size_t)GB * 64 * 64 * GC * 2));
  unsigned short* w1   = (unsigned short*)(ws + take(1024 * 2));
  unsigned short* w2   = (unsigned short*)(ws + take(9216 * 2));
  unsigned short* w3   = (unsigned short*)(ws + take(9216 * 2));
  unsigned short* w4   = (unsigned short*)(ws + take(9216 * 2));

  // grid + m1 are contiguous (sizes are 256-multiples): zero both in one pass.
  long n16 = (long)(((size_t)GB * GG * GG * 2 + (size_t)GB * GG * GG) / 16);
  k_zero<<<1024, 256, 0, stream>>>((v4u*)grid, n16);
  k_prep<<<112, 256, 0, stream>>>(K1, K2, K3, K4, w1, w2, w3, w4);
  k_scatter<<<(GB * 32768 + 255) / 256, 256, 0, stream>>>(coords, feats, grid, m1, GB * 32768);
  k_conv1<<<dim3(4, 512, 8), 256, 0, stream>>>(grid, m1, w1, y1);
  k_down<<<(GB * 256 * 256 + 255) / 256, 256, 0, stream>>>(m1, m2, 256, 256);
  k_conv_s2<<<dim3(4, 256, 8), 128, 0, stream>>>(y1, y2, m2, w2, 512, 512, 256, 256);
  k_down<<<(GB * 128 * 128 + 255) / 256, 256, 0, stream>>>(m2, m3, 128, 128);
  k_conv_s2<<<dim3(2, 128, 8), 128, 0, stream>>>(y2, y3, m3, w3, 256, 256, 128, 128);
  k_down<<<(GB * 64 * 64 + 255) / 256, 256, 0, stream>>>(m3, m4, 64, 64);
  k_conv_s2<<<dim3(1, 64, 8), 128, 0, stream>>>(y3, y4, m4, w4, 128, 128, 64, 64);
  k_head<<<8, 256, 0, stream>>>(y4, m4, x2, W1, b1, W2, b2, W3, b3, out);
}
